// Transformer_partial_structure_12103217840228
// MI455X (gfx1250) — matrix operand order repaired and checked by reading, compiled, not run on hardware
//
#include <hip/hip_runtime.h>
#include <hip/hip_bf16.h>

// ---------------------------------------------------------------------------
// Types for gfx1250 WMMA (wave32): v_wmma_f32_16x16x32_bf16
// ---------------------------------------------------------------------------
typedef __bf16 bf16_t;
typedef __attribute__((ext_vector_type(16))) __bf16 v16bf;
typedef __attribute__((ext_vector_type(8)))  __bf16 v8bf;
typedef __attribute__((ext_vector_type(8)))  float  v8f;

// Model constants
#define BB    8
#define NTOK  2048
#define DIM   768
#define HEADS 12
#define DH    64
#define MLPD  3072
#define DEPTH 4
#define LM    64
#define KC    33
#define TOK   (BB * NTOK)      // 16384 rows

enum { F_BIAS = 1, F_GELU = 2, F_RESID = 4, F_F32 = 8, F_BF16 = 16, F_BF16T = 32 };

struct BStrides {
    long long a_b, a_h, b_b, b_h, c_b, c_h, cb_b, cb_h, r_b, r_h;
};

// Build one 16-element bf16 fragment from two contiguous 16B runs.
__device__ __forceinline__ v16bf ld_frag_pair(const bf16_t* p0, const bf16_t* p1)
{
    const v8bf lo = *reinterpret_cast<const v8bf*>(p0);
    const v8bf hi = *reinterpret_cast<const v8bf*>(p1);
    return __builtin_shufflevector(lo, hi, 0, 1, 2, 3, 4, 5, 6, 7,
                                   8, 9, 10, 11, 12, 13, 14, 15);
}

// ---------------------------------------------------------------------------
// Batched bf16 GEMM: C[M,N] = act( scale * A[M,K] @ B[K,N] + bias ) + resid
// A: M x K row-major (lda). B supplied TRANSPOSED: N x K row-major (ldb).
// 128 threads = 4 waves; each wave computes a 32x64 strip (block tile 128x64).
// Fragments loaded straight from global memory as b128 pairs (no LDS).
// ---------------------------------------------------------------------------
__global__ __launch_bounds__(128) void gemm_kernel(
    const bf16_t* __restrict__ A, int lda,
    const bf16_t* __restrict__ Bm, int ldb,
    float* __restrict__ C, int ldc,
    bf16_t* __restrict__ Cb, int ldcb,
    const float* __restrict__ bias,
    const float* __restrict__ resid, int ldr,
    int M, int N, int K, float scale, int flags, int batchH, BStrides bs)
{
    const int zb   = blockIdx.z;
    const int bidx = zb / batchH;
    const int hidx = zb % batchH;
    A     += (long long)bidx * bs.a_b  + (long long)hidx * bs.a_h;
    Bm    += (long long)bidx * bs.b_b  + (long long)hidx * bs.b_h;
    C     += (long long)bidx * bs.c_b  + (long long)hidx * bs.c_h;
    Cb    += (long long)bidx * bs.cb_b + (long long)hidx * bs.cb_h;
    resid += (long long)bidx * bs.r_b  + (long long)hidx * bs.r_h;

    const int lane = threadIdx.x & 31;
    const int wave = threadIdx.x >> 5;
    const long long rowBase = (long long)blockIdx.y * 128 + wave * 32;
    if (rowBase >= M) return;                 // uniform per wave
    const long long colBase = (long long)blockIdx.x * 64;

    const int l15  = lane & 15;
    const int kbA  = (lane >> 4) * 8;         // A prefetch offset only; A loads use kb16
    const int kb16 = (lane >> 4) * 16;        // B: K run  at kb16 (16 long)

    const bf16_t* arow0 = A + (rowBase + l15) * (long long)lda;
    const bf16_t* arow1 = A + (rowBase + 16 + l15) * (long long)lda;
    const bf16_t* brow0 = Bm + (colBase + 0  + l15) * (long long)ldb;
    const bf16_t* brow1 = Bm + (colBase + 16 + l15) * (long long)ldb;
    const bf16_t* brow2 = Bm + (colBase + 32 + l15) * (long long)ldb;
    const bf16_t* brow3 = Bm + (colBase + 48 + l15) * (long long)ldb;

    v8f acc0[4], acc1[4];
#pragma unroll
    for (int c = 0; c < 4; ++c)
#pragma unroll
        for (int e = 0; e < 8; ++e) { acc0[c][e] = 0.0f; acc1[c][e] = 0.0f; }

    for (int k0 = 0; k0 < K; k0 += 32) {
        if (k0 + 32 < K) {
            __builtin_prefetch(arow0 + k0 + 32 + kbA, 0, 0);
            __builtin_prefetch(brow0 + k0 + 32 + kb16, 0, 0);
        }
        const v16bf a0 = ld_frag_pair(arow0 + k0 + kb16, arow0 + k0 + kb16 + 8);
        const v16bf a1 = ld_frag_pair(arow1 + k0 + kb16, arow1 + k0 + kb16 + 8);

        const v16bf b0 = ld_frag_pair(brow0 + k0 + kb16, brow0 + k0 + kb16 + 8);
        acc0[0] = __builtin_amdgcn_wmma_f32_16x16x32_bf16(false, a0, false, b0, (short)0, acc0[0], false, false);
        acc1[0] = __builtin_amdgcn_wmma_f32_16x16x32_bf16(false, a1, false, b0, (short)0, acc1[0], false, false);

        const v16bf b1 = ld_frag_pair(brow1 + k0 + kb16, brow1 + k0 + kb16 + 8);
        acc0[1] = __builtin_amdgcn_wmma_f32_16x16x32_bf16(false, a0, false, b1, (short)0, acc0[1], false, false);
        acc1[1] = __builtin_amdgcn_wmma_f32_16x16x32_bf16(false, a1, false, b1, (short)0, acc1[1], false, false);

        const v16bf b2 = ld_frag_pair(brow2 + k0 + kb16, brow2 + k0 + kb16 + 8);
        acc0[2] = __builtin_amdgcn_wmma_f32_16x16x32_bf16(false, a0, false, b2, (short)0, acc0[2], false, false);
        acc1[2] = __builtin_amdgcn_wmma_f32_16x16x32_bf16(false, a1, false, b2, (short)0, acc1[2], false, false);

        const v16bf b3 = ld_frag_pair(brow3 + k0 + kb16, brow3 + k0 + kb16 + 8);
        acc0[3] = __builtin_amdgcn_wmma_f32_16x16x32_bf16(false, a0, false, b3, (short)0, acc0[3], false, false);
        acc1[3] = __builtin_amdgcn_wmma_f32_16x16x32_bf16(false, a1, false, b3, (short)0, acc1[3], false, false);
    }

    // --- epilogue (ISA 32-bit C/D 16x16 layout) ---------------------------
    const int ncol = lane & 15;
    const int mhi  = (lane >> 4) * 8;
#pragma unroll
    for (int g = 0; g < 2; ++g) {
#pragma unroll
        for (int c = 0; c < 4; ++c) {
#pragma unroll
            for (int v = 0; v < 8; ++v) {
                const long long row = rowBase + g * 16 + mhi + v;
                const long long col = colBase + c * 16 + ncol;
                float val = (g ? acc1[c][v] : acc0[c][v]) * scale;
                if (flags & F_BIAS)  val += bias[col];
                if (flags & F_GELU)  val = 0.5f * val * (1.0f + erff(val * 0.70710678118654752f));
                if (flags & F_RESID) val += resid[row * ldr + col];
                if (flags & F_F32)   C[row * ldc + col] = val;
                if (flags & F_BF16)  Cb[row * ldcb + col] = (bf16_t)val;
                if (flags & F_BF16T) Cb[col * ldcb + row] = (bf16_t)val;
            }
        }
    }
}

// ---------------------------------------------------------------------------
// Transpose-convert weights: in (L, K, N) f32 -> out (L, N, K) bf16
// ---------------------------------------------------------------------------
__global__ __launch_bounds__(256) void cvtT_kernel(const float* __restrict__ in,
                                                   bf16_t* __restrict__ out,
                                                   int L, int K, int N)
{
    const long long idx = (long long)blockIdx.x * 256 + threadIdx.x;
    const long long total = (long long)L * K * N;
    if (idx >= total) return;
    const long long kn = (long long)K * N;
    const int l = (int)(idx / kn);
    const long long rem = idx - (long long)l * kn;
    const int k = (int)(rem / N);
    const int n = (int)(rem - (long long)k * N);
    out[(long long)l * kn + (long long)n * K + k] = (bf16_t)in[idx];
}

// ---------------------------------------------------------------------------
// v^T per head: vt[b,h,d,n] = kv[b*2048+n, 768 + h*64 + d]  (f32 -> bf16)
// ---------------------------------------------------------------------------
__global__ __launch_bounds__(256) void vtrans_kernel(const float* __restrict__ kvf,
                                                     bf16_t* __restrict__ vt)
{
    const long long idx = (long long)blockIdx.x * 256 + threadIdx.x;
    const long long total = (long long)BB * HEADS * DH * NTOK;
    if (idx >= total) return;
    const int n = (int)(idx & (NTOK - 1));
    const int d = (int)((idx >> 11) & 63);
    const int bh = (int)(idx >> 17);
    const int h = bh % HEADS, b = bh / HEADS;
    vt[idx] = (bf16_t)kvf[((long long)b * NTOK + n) * (2 * DIM) + DIM + h * DH + d];
}

// ---------------------------------------------------------------------------
// Patch rearrange + LayerNorm(128) -> bf16 tokens.
// b p c (f pf) (h p1) (w p2) -> b (p f h w) (c p1 p2 pf)
// ---------------------------------------------------------------------------
__global__ __launch_bounds__(128) void embed_tok_kernel(
    const float* __restrict__ ps, const float* __restrict__ g,
    const float* __restrict__ bta, bf16_t* __restrict__ out)
{
    const int row = blockIdx.x;          // b*2048 + t
    const int b = row >> 11, t = row & 2047;
    const int d = threadIdx.x;           // 0..127
    const int wi = t & 7, hi = (t >> 3) & 7, fi = (t >> 6) & 3, p = (t >> 8) & 7;
    const int pf = d & 3, p2 = (d >> 2) & 3, p1 = (d >> 4) & 3, c = (d >> 6) & 1;
    const long long src = (long long)b * 262144 + p * 32768 + c * 16384
                        + (fi * 4 + pf) * 1024 + (hi * 4 + p1) * 32 + (wi * 4 + p2);
    const float v = ps[src];

    __shared__ float red[128];
    red[d] = v; __syncthreads();
    for (int o = 64; o > 0; o >>= 1) { if (d < o) red[d] += red[d + o]; __syncthreads(); }
    const float mu = red[0] * (1.0f / 128.0f); __syncthreads();
    const float dv = v - mu;
    red[d] = dv * dv; __syncthreads();
    for (int o = 64; o > 0; o >>= 1) { if (d < o) red[d] += red[d + o]; __syncthreads(); }
    const float rs = rsqrtf(red[0] * (1.0f / 128.0f) + 1e-5f);
    out[(long long)row * 128 + d] = (bf16_t)(dv * rs * g[d] + bta[d]);
}

// ---------------------------------------------------------------------------
// Generic LayerNorm over last dim D (block per row). Optional +pos, dual out.
// ---------------------------------------------------------------------------
__global__ __launch_bounds__(256) void ln_kernel(
    const float* __restrict__ in, const float* __restrict__ g,
    const float* __restrict__ bta, const float* __restrict__ pos, int posmod,
    float* __restrict__ outf, bf16_t* __restrict__ outb, int D)
{
    const long long row = blockIdx.x;
    const float* x = in + row * D;
    __shared__ float red[256];
    const int tid = threadIdx.x;

    float s = 0.0f;
    for (int d = tid; d < D; d += 256) s += x[d];
    red[tid] = s; __syncthreads();
    for (int o = 128; o > 0; o >>= 1) { if (tid < o) red[tid] += red[tid + o]; __syncthreads(); }
    const float mu = red[0] / (float)D; __syncthreads();

    float s2 = 0.0f;
    for (int d = tid; d < D; d += 256) { const float t = x[d] - mu; s2 += t * t; }
    red[tid] = s2; __syncthreads();
    for (int o = 128; o > 0; o >>= 1) { if (tid < o) red[tid] += red[tid + o]; __syncthreads(); }
    const float rs = rsqrtf(red[0] / (float)D + 1e-5f);

    for (int d = tid; d < D; d += 256) {
        float v = (x[d] - mu) * rs * g[d] + bta[d];
        if (pos)  v += pos[(long long)(row % posmod) * D + d];
        if (outf) outf[row * D + d] = v;
        if (outb) outb[row * D + d] = (bf16_t)v;
    }
}

// ---------------------------------------------------------------------------
// Landmark means: q_l/k_l[b,h,m,d] = mean_{li<32} of q/k rows (bf16 out).
// ---------------------------------------------------------------------------
__global__ __launch_bounds__(256) void landmark_kernel(
    const float* __restrict__ q, const float* __restrict__ kv,
    bf16_t* __restrict__ qlb, bf16_t* __restrict__ klb)
{
    const int idx = blockIdx.x * 256 + threadIdx.x;       // B*H*64*64
    if (idx >= BB * HEADS * LM * DH) return;
    const int d = idx & 63, m = (idx >> 6) & 63;
    const int bh = idx >> 12, h = bh % HEADS, b = bh / HEADS;
    float sq = 0.0f, sk = 0.0f;
    for (int li = 0; li < 32; ++li) {
        const long long row = (long long)b * NTOK + m * 32 + li;
        sq += q[row * DIM + h * DH + d];
        sk += kv[row * (2 * DIM) + h * DH + d];
    }
    qlb[idx] = (bf16_t)(sq * (1.0f / 32.0f));
    klb[idx] = (bf16_t)(sk * (1.0f / 32.0f));
}

// ---------------------------------------------------------------------------
// Row softmax (block per row), dual f32 / bf16 output. C <= 2048.
// ---------------------------------------------------------------------------
__global__ __launch_bounds__(256) void softmax_kernel(
    const float* __restrict__ in, float* __restrict__ outf,
    bf16_t* __restrict__ outb, int C)
{
    const long long base = (long long)blockIdx.x * C;
    const int tid = threadIdx.x;
    __shared__ float red[256];

    float mx = -3.4e38f;
    for (int c = tid; c < C; c += 256) mx = fmaxf(mx, in[base + c]);
    red[tid] = mx; __syncthreads();
    for (int o = 128; o > 0; o >>= 1) { if (tid < o) red[tid] = fmaxf(red[tid], red[tid + o]); __syncthreads(); }
    mx = red[0]; __syncthreads();

    float loc[8];
    float s = 0.0f;
    int cnt = 0;
    for (int c = tid; c < C; c += 256) { const float e = __expf(in[base + c] - mx); loc[cnt++] = e; s += e; }
    red[tid] = s; __syncthreads();
    for (int o = 128; o > 0; o >>= 1) { if (tid < o) red[tid] += red[tid + o]; __syncthreads(); }
    const float inv = 1.0f / red[0];

    cnt = 0;
    for (int c = tid; c < C; c += 256) {
        const float v = loc[cnt++] * inv;
        outf[base + c] = v;
        if (outb) outb[base + c] = (bf16_t)v;
    }
}

// ---------------------------------------------------------------------------
// Moore-Penrose iterative pseudo-inverse of 64x64 (per b,h), 6 iterations.
// ---------------------------------------------------------------------------
__device__ inline void mm64_regs(const float* __restrict__ A,
                                 const float* __restrict__ Bm,
                                 float* __restrict__ reg, int tid)
{
#pragma unroll
    for (int j = 0; j < 16; ++j) {
        const int t = tid + j * 256;
        const int r = t >> 6, c = t & 63;
        float s = 0.0f;
        for (int k = 0; k < 64; ++k) s += A[r * 64 + k] * Bm[k * 64 + c];
        reg[j] = s;
    }
}

__global__ __launch_bounds__(256) void pinv_kernel(
    const float* __restrict__ a2, bf16_t* __restrict__ zb)
{
    __shared__ float Xs[4096], Zs[4096], Ps[4096], Qs[4096];
    const int bh = blockIdx.x, tid = threadIdx.x;
    const float* X = a2 + (long long)bh * 4096;

    for (int t = tid; t < 4096; t += 256) Xs[t] = X[t];
    __syncthreads();
    if (tid < 64) {            // row sums of |X|
        float s = 0.0f;
        for (int c = 0; c < 64; ++c) s += fabsf(Xs[tid * 64 + c]);
        Ps[tid] = s;
    } else if (tid < 128) {    // column sums of |X|
        const int c = tid - 64; float s = 0.0f;
        for (int r = 0; r < 64; ++r) s += fabsf(Xs[r * 64 + c]);
        Ps[tid] = s;
    }
    __syncthreads();
    if (tid == 0) {
        float mr = 0.0f, mc = 0.0f;
        for (int r = 0; r < 64; ++r)   mr = fmaxf(mr, Ps[r]);
        for (int c = 64; c < 128; ++c) mc = fmaxf(mc, Ps[c]);
        Ps[128] = 1.0f / (mr * mc);
    }
    __syncthreads();
    const float invn = Ps[128];
    __syncthreads();
    for (int t = tid; t < 4096; t += 256) {
        const int r = t >> 6, c = t & 63;
        Zs[t] = Xs[c * 64 + r] * invn;   // X^T / (col*row)
    }
    __syncthreads();

    float reg[16];
    for (int it = 0; it < 6; ++it) {
        mm64_regs(Xs, Zs, reg, tid); __syncthreads();            // P = X@Z
#pragma unroll
        for (int j = 0; j < 16; ++j) Ps[tid + j * 256] = reg[j];
        __syncthreads();
        for (int t = tid; t < 4096; t += 256) {                  // Q = 7I - P
            const int r = t >> 6, c = t & 63;
            Qs[t] = ((r == c) ? 7.0f : 0.0f) - Ps[t];
        }
        __syncthreads();
        mm64_regs(Ps, Qs, reg, tid); __syncthreads();            // Q = 15I - P@Q
#pragma unroll
        for (int j = 0; j < 16; ++j) {
            const int t = tid + j * 256, r = t >> 6, c = t & 63;
            Qs[t] = ((r == c) ? 15.0f : 0.0f) - reg[j];
        }
        __syncthreads();
        mm64_regs(Ps, Qs, reg, tid); __syncthreads();            // Q = 13I - P@Q
#pragma unroll
        for (int j = 0; j < 16; ++j) {
            const int t = tid + j * 256, r = t >> 6, c = t & 63;
            Qs[t] = ((r == c) ? 13.0f : 0.0f) - reg[j];
        }
        __syncthreads();
        mm64_regs(Zs, Qs, reg, tid); __syncthreads();            // Z = 0.25*Z@Q
#pragma unroll
        for (int j = 0; j < 16; ++j) Zs[tid + j * 256] = 0.25f * reg[j];
        __syncthreads();
    }
    for (int t = tid; t < 4096; t += 256)
        zb[(long long)bh * 4096 + t] = (bf16_t)Zs[t];
}

// ---------------------------------------------------------------------------
// Depthwise conv over sequence dim of v (kernel 33, pad 16, groups = heads).
// v lives in kv f32 at column offset 768. Output cv: (B, N, 768).
// ---------------------------------------------------------------------------
__global__ __launch_bounds__(256) void dwconv_kernel(
    const float* __restrict__ kv, const float* __restrict__ cw,
    float* __restrict__ cv)
{
    const long long idx = (long long)blockIdx.x * 256 + threadIdx.x;
    const long long total = (long long)TOK * DIM;
    if (idx >= total) return;
    const int col = (int)(idx % DIM);
    const long long rown = idx / DIM;
    const int n = (int)(rown % NTOK);
    const int b = (int)(rown / NTOK);
    const int h = col >> 6;
    float s = 0.0f;
    for (int t = 0; t < KC; ++t) {
        const int nn = n + t - (KC / 2);
        if ((unsigned)nn < (unsigned)NTOK)
            s += cw[h * KC + t] * kv[((long long)b * NTOK + nn) * (2 * DIM) + DIM + col];
    }
    cv[idx] = s;
}

// ---------------------------------------------------------------------------
// Host-side GEMM launcher
// ---------------------------------------------------------------------------
struct GArg {
    const bf16_t* A; int lda; long long a_b, a_h;
    const bf16_t* B; int ldb; long long b_b, b_h;   // B is N x K (transposed)
    float* C; int ldc; long long c_b, c_h;
    bf16_t* Cb; int ldcb; long long cb_b, cb_h;
    const float* bias;
    const float* R; int ldr; long long r_b, r_h;
    int M, N, K; float scale; int flags; int batch; int batchH;
};

static void run_gemm(const GArg& g, hipStream_t st)
{
    dim3 grid((unsigned)(g.N / 64), (unsigned)((g.M + 127) / 128), (unsigned)g.batch);
    BStrides bs{g.a_b, g.a_h, g.b_b, g.b_h, g.c_b, g.c_h, g.cb_b, g.cb_h, g.r_b, g.r_h};
    gemm_kernel<<<grid, 128, 0, st>>>(g.A, g.lda, g.B, g.ldb, g.C, g.ldc,
        g.Cb, g.ldcb, g.bias, g.R, g.ldr, g.M, g.N, g.K, g.scale, g.flags, g.batchH, bs);
}

static void run_cvtT(const float* in, bf16_t* out, int L, int K, int N, hipStream_t st)
{
    const long long n = (long long)L * K * N;
    cvtT_kernel<<<(unsigned)((n + 255) / 256), 256, 0, st>>>(in, out, L, K, N);
}

// ---------------------------------------------------------------------------
extern "C" void kernel_launch(void* const* d_in, const int* in_sizes, int n_in,
                              void* d_out, int out_size, void* d_ws, size_t ws_size,
                              hipStream_t stream)
{
    (void)in_sizes; (void)n_in; (void)out_size; (void)ws_size;

    const float* x_in  = (const float*)d_in[0];
    const float* pstr  = (const float*)d_in[1];
    const float* pos   = (const float*)d_in[2];
    const float* eln1g = (const float*)d_in[3];
    const float* eln1b = (const float*)d_in[4];
    const float* emb_w = (const float*)d_in[5];
    const float* emb_b = (const float*)d_in[6];
    const float* eln2g = (const float*)d_in[7];
    const float* eln2b = (const float*)d_in[8];
    const float* lnxg  = (const float*)d_in[9];
    const float* lnxb  = (const float*)d_in[10];
    const float* lnpg  = (const float*)d_in[11];
    const float* lnpb  = (const float*)d_in[12];
    const float* wq    = (const float*)d_in[13];
    const float* wkv   = (const float*)d_in[14];
    const float* convw = (const float*)d_in[15];
    const float* wo    = (const float*)d_in[16];
    const float* bo    = (const float*)d_in[17];
    const float* ffg   = (const float*)d_in[18];
    const float* ffb   = (const float*)d_in[19];
    const float* w1    = (const float*)d_in[20];
    const float* b1    = (const float*)d_in[21];
    const float* w2    = (const float*)d_in[22];
    const float* b2    = (const float*)d_in[23];
    float* xcur = (float*)d_out;

    // ---- workspace carve ------------------------------------------------
    size_t off = 0;
    auto alloc = [&](size_t bytes) -> void* {
        void* p = (char*)d_ws + off;
        off = (off + bytes + 255) & ~(size_t)255;
        return p;
    };
    const long long BHLL = (long long)BB * HEADS * LM * DH;   // 393216

    // transposed bf16 weights (N x K per layer)
    bf16_t* emb_wT = (bf16_t*)alloc(128LL * DIM * 2);
    bf16_t* wqT    = (bf16_t*)alloc((long long)DEPTH * DIM * DIM * 2);
    bf16_t* wkvT   = (bf16_t*)alloc((long long)DEPTH * DIM * 2 * DIM * 2);
    bf16_t* woT    = (bf16_t*)alloc((long long)DEPTH * DIM * DIM * 2);
    bf16_t* w1T    = (bf16_t*)alloc((long long)DEPTH * DIM * MLPD * 2);
    bf16_t* w2T    = (bf16_t*)alloc((long long)DEPTH * MLPD * DIM * 2);

    bf16_t* tok_bf = (bf16_t*)alloc((long long)TOK * 128 * 2);
    float*  psb    = (float*) alloc((long long)TOK * DIM * 4);
    bf16_t* xn_bf  = (bf16_t*)alloc((long long)TOK * DIM * 2);
    bf16_t* pn_bf  = (bf16_t*)alloc((long long)TOK * DIM * 2);
    float*  qf     = (float*) alloc((long long)TOK * DIM * 4);
    bf16_t* q_bf   = (bf16_t*)alloc((long long)TOK * DIM * 2);
    float*  kvf    = (float*) alloc((long long)TOK * 2 * DIM * 4);
    bf16_t* kv_bf  = (bf16_t*)alloc((long long)TOK * 2 * DIM * 2);
    bf16_t* vt_bf  = (bf16_t*)alloc((long long)BB * HEADS * DH * NTOK * 2);
    bf16_t* ql_bf  = (bf16_t*)alloc(BHLL * 2);
    bf16_t* kl_bf  = (bf16_t*)alloc(BHLL * 2);
    float*  attn1  = (float*) alloc((long long)BB * HEADS * NTOK * LM * 4);
    bf16_t* a1_bf  = (bf16_t*)alloc((long long)BB * HEADS * NTOK * LM * 2);
    float*  attn2  = (float*) alloc(BHLL * 4);
    float*  attn3  = (float*) alloc((long long)BB * HEADS * LM * NTOK * 4);
    bf16_t* a3_bf  = (bf16_t*)alloc((long long)BB * HEADS * LM * NTOK * 2);
    bf16_t* z_bf   = (bf16_t*)alloc(BHLL * 2);
    bf16_t* w2mT   = (bf16_t*)alloc(BHLL * 2);
    bf16_t* zwT    = (bf16_t*)alloc(BHLL * 2);
    float*  cv     = (float*) alloc((long long)TOK * DIM * 4);
    bf16_t* ao_bf  = (bf16_t*)alloc((long long)TOK * DIM * 2);
    bf16_t* h_bf   = (bf16_t*)alloc((long long)TOK * MLPD * 2);

    // ---- weights -> bf16, transposed to N x K ---------------------------
    run_cvtT(emb_w, emb_wT, 1, 128, DIM, stream);
    run_cvtT(wq,  wqT,  DEPTH, DIM, DIM, stream);
    run_cvtT(wkv, wkvT, DEPTH, DIM, 2 * DIM, stream);
    run_cvtT(wo,  woT,  DEPTH, DIM, DIM, stream);
    run_cvtT(w1,  w1T,  DEPTH, DIM, MLPD, stream);
    run_cvtT(w2,  w2T,  DEPTH, MLPD, DIM, stream);

    // ---- running x lives in d_out ---------------------------------------
    hipMemcpyAsync(xcur, x_in, (size_t)TOK * DIM * sizeof(float),
                   hipMemcpyDeviceToDevice, stream);

    // ---- patch embedding -------------------------------------------------
    embed_tok_kernel<<<TOK, 128, 0, stream>>>(pstr, eln1g, eln1b, tok_bf);
    {
        GArg g{}; g.A = tok_bf; g.lda = 128; g.B = emb_wT; g.ldb = 128;
        g.C = psb; g.ldc = DIM; g.bias = emb_b;
        g.M = TOK; g.N = DIM; g.K = 128; g.scale = 1.0f;
        g.flags = F_BIAS | F_F32; g.batch = 1; g.batchH = 1;
        run_gemm(g, stream);
    }
    ln_kernel<<<TOK, 256, 0, stream>>>(psb, eln2g, eln2b, pos, NTOK, psb, nullptr, DIM);

    // ---- transformer layers ----------------------------------------------
    const long long s_q_b  = (long long)NTOK * DIM;
    const long long s_kv_b = (long long)NTOK * 2 * DIM;
    const long long s_lm_h = (long long)LM * DH;             // 4096
    const long long s_lm_b = (long long)HEADS * s_lm_h;
    const long long s_a1_h = (long long)NTOK * LM;
    const long long s_a1_b = (long long)HEADS * s_a1_h;
    const long long s_a3_h = (long long)LM * NTOK;
    const long long s_a3_b = (long long)HEADS * s_a3_h;
    const long long s_vt_h = (long long)DH * NTOK;
    const long long s_vt_b = (long long)HEADS * s_vt_h;

    for (int i = 0; i < DEPTH; ++i) {
        const bf16_t* wqT_i  = wqT  + (long long)i * DIM * DIM;
        const bf16_t* wkvT_i = wkvT + (long long)i * DIM * 2 * DIM;
        const bf16_t* woT_i  = woT  + (long long)i * DIM * DIM;
        const bf16_t* w1T_i  = w1T  + (long long)i * DIM * MLPD;
        const bf16_t* w2T_i  = w2T  + (long long)i * MLPD * DIM;

        // LN(x) -> xn_bf ; LN(ps) -> pn_bf
        ln_kernel<<<TOK, 256, 0, stream>>>(xcur, lnxg + i * DIM, lnxb + i * DIM,
                                           nullptr, 1, nullptr, xn_bf, DIM);
        ln_kernel<<<TOK, 256, 0, stream>>>(psb, lnpg + i * DIM, lnpb + i * DIM,
                                           nullptr, 1, nullptr, pn_bf, DIM);

        // q = LN(x) @ wq * DH^-0.5
        { GArg g{}; g.A = xn_bf; g.lda = DIM; g.B = wqT_i; g.ldb = DIM;
          g.C = qf; g.ldc = DIM; g.Cb = q_bf; g.ldcb = DIM;
          g.M = TOK; g.N = DIM; g.K = DIM; g.scale = 0.125f;
          g.flags = F_F32 | F_BF16; g.batch = 1; g.batchH = 1; run_gemm(g, stream); }
        // kv = LN(ps) @ wkv
        { GArg g{}; g.A = pn_bf; g.lda = DIM; g.B = wkvT_i; g.ldb = DIM;
          g.C = kvf; g.ldc = 2 * DIM; g.Cb = kv_bf; g.ldcb = 2 * DIM;
          g.M = TOK; g.N = 2 * DIM; g.K = DIM; g.scale = 1.0f;
          g.flags = F_F32 | F_BF16; g.batch = 1; g.batchH = 1; run_gemm(g, stream); }

        // landmarks + v^T
        landmark_kernel<<<(BB * HEADS * LM * DH + 255) / 256, 256, 0, stream>>>(
            qf, kvf, ql_bf, kl_bf);
        vtrans_kernel<<<(unsigned)(((long long)BB * HEADS * DH * NTOK + 255) / 256),
                        256, 0, stream>>>(kvf, vt_bf);

        // attn1 logits = q @ k_l^T   (B operand = k_l, naturally N x K)
        { GArg g{}; g.A = q_bf; g.lda = DIM; g.a_b = s_q_b; g.a_h = DH;
          g.B = kl_bf; g.ldb = DH; g.b_b = s_lm_b; g.b_h = s_lm_h;
          g.C = attn1; g.ldc = LM; g.c_b = s_a1_b; g.c_h = s_a1_h;
          g.M = NTOK; g.N = LM; g.K = DH; g.scale = 1.0f;
          g.flags = F_F32; g.batch = BB * HEADS; g.batchH = HEADS; run_gemm(g, stream); }
        // attn2 logits = q_l @ k_l^T
        { GArg g{}; g.A = ql_bf; g.lda = DH; g.a_b = s_lm_b; g.a_h = s_lm_h;
          g.B = kl_bf; g.ldb = DH; g.b_b = s_lm_b; g.b_h = s_lm_h;
          g.C = attn2; g.ldc = DH; g.c_b = s_lm_b; g.c_h = s_lm_h;
          g.M = LM; g.N = LM; g.K = DH; g.scale = 1.0f;
          g.flags = F_F32; g.batch = BB * HEADS; g.batchH = HEADS; run_gemm(g, stream); }
        // attn3 logits = q_l @ k^T   (B operand = k rows inside kv_bf)
        { GArg g{}; g.A = ql_bf; g.lda = DH; g.a_b = s_lm_b; g.a_h = s_lm_h;
          g.B = kv_bf; g.ldb = 2 * DIM; g.b_b = s_kv_b; g.b_h = DH;
          g.C = attn3; g.ldc = NTOK; g.c_b = s_a3_b; g.c_h = s_a3_h;
          g.M = LM; g.N = NTOK; g.K = DH; g.scale = 1.0f;
          g.flags = F_F32; g.batch = BB * HEADS; g.batchH = HEADS; run_gemm(g, stream); }

        // softmaxes
        softmax_kernel<<<BB * HEADS * NTOK, 256, 0, stream>>>(attn1, attn1, a1_bf, LM);
        softmax_kernel<<<BB * HEADS * LM, 256, 0, stream>>>(attn2, attn2, nullptr, LM);
        softmax_kernel<<<BB * HEADS * LM, 256, 0, stream>>>(attn3, attn3, a3_bf, NTOK);

        // z = mp_pinv(attn2)
        pinv_kernel<<<BB * HEADS, 256, 0, stream>>>(attn2, z_bf);

        // w2m = attn3 @ v  (B operand = v^T), store transposed -> w2mT
        { GArg g{}; g.A = a3_bf; g.lda = NTOK; g.a_b = s_a3_b; g.a_h = s_a3_h;
          g.B = vt_bf; g.ldb = NTOK; g.b_b = s_vt_b; g.b_h = s_vt_h;
          g.Cb = w2mT; g.ldcb = LM; g.cb_b = s_lm_b; g.cb_h = s_lm_h;
          g.M = LM; g.N = DH; g.K = NTOK; g.scale = 1.0f;
          g.flags = F_BF16T; g.batch = BB * HEADS; g.batchH = HEADS; run_gemm(g, stream); }
        // zw = z @ w2m  (B operand = w2mT), store transposed -> zwT
        { GArg g{}; g.A = z_bf; g.lda = LM; g.a_b = s_lm_b; g.a_h = s_lm_h;
          g.B = w2mT; g.ldb = LM; g.b_b = s_lm_b; g.b_h = s_lm_h;
          g.Cb = zwT; g.ldcb = LM; g.cb_b = s_lm_b; g.cb_h = s_lm_h;
          g.M = LM; g.N = DH; g.K = LM; g.scale = 1.0f;
          g.flags = F_BF16T; g.batch = BB * HEADS; g.batchH = HEADS; run_gemm(g, stream); }

        // depthwise conv of v over sequence -> cv (B,N,768)
        dwconv_kernel<<<(unsigned)(((long long)TOK * DIM + 255) / 256), 256, 0, stream>>>(
            kvf, convw + (long long)i * HEADS * KC, cv);

        // attn_out = attn1 @ zw + conv  (B operand = zwT) -> ao_bf (B,N,768)
        { GArg g{}; g.A = a1_bf; g.lda = LM; g.a_b = s_a1_b; g.a_h = s_a1_h;
          g.B = zwT; g.ldb = LM; g.b_b = s_lm_b; g.b_h = s_lm_h;
          g.Cb = ao_bf; g.ldcb = DIM; g.cb_b = s_q_b; g.cb_h = DH;
          g.R = cv; g.ldr = DIM; g.r_b = s_q_b; g.r_h = DH;
          g.M = NTOK; g.N = DH; g.K = LM; g.scale = 1.0f;
          g.flags = F_RESID | F_BF16; g.batch = BB * HEADS; g.batchH = HEADS; run_gemm(g, stream); }

        // x = attn_out @ wo + bo + x
        { GArg g{}; g.A = ao_bf; g.lda = DIM; g.B = woT_i; g.ldb = DIM;
          g.C = xcur; g.ldc = DIM; g.bias = bo + i * DIM;
          g.R = xcur; g.ldr = DIM;
          g.M = TOK; g.N = DIM; g.K = DIM; g.scale = 1.0f;
          g.flags = F_BIAS | F_RESID | F_F32; g.batch = 1; g.batchH = 1; run_gemm(g, stream); }

        // MLP
        ln_kernel<<<TOK, 256, 0, stream>>>(xcur, ffg + i * DIM, ffb + i * DIM,
                                           nullptr, 1, nullptr, xn_bf, DIM);
        { GArg g{}; g.A = xn_bf; g.lda = DIM; g.B = w1T_i; g.ldb = DIM;
          g.Cb = h_bf; g.ldcb = MLPD; g.bias = b1 + i * MLPD;
          g.M = TOK; g.N = MLPD; g.K = DIM; g.scale = 1.0f;
          g.flags = F_BIAS | F_GELU | F_BF16; g.batch = 1; g.batchH = 1; run_gemm(g, stream); }
        { GArg g{}; g.A = h_bf; g.lda = MLPD; g.B = w2T_i; g.ldb = MLPD;
          g.C = xcur; g.ldc = DIM; g.bias = b2 + i * DIM;
          g.R = xcur; g.ldr = DIM;
          g.M = TOK; g.N = DIM; g.K = MLPD; g.scale = 1.0f;
          g.flags = F_BIAS | F_RESID | F_F32; g.batch = 1; g.batchH = 1; run_gemm(g, stream); }
    }
}
